// AttentionBlockDraft_317827580344
// MI455X (gfx1250) — matrix operand order repaired and checked by reading, not compiled, not run on hardware
//
#include <hip/hip_runtime.h>

typedef __attribute__((ext_vector_type(16))) __bf16 bf16x16;
typedef __attribute__((ext_vector_type(8)))  __bf16 bf16x8;
typedef __attribute__((ext_vector_type(8)))  float  f32x8;
typedef __attribute__((ext_vector_type(4)))  float  f32x4;

#define F_TOT 224
#define PADW  232   // padded LDS row stride (bf16 elements), 16B-aligned rows

static __device__ __forceinline__ f32x8 wmma_bf16(bf16x16 a, bf16x16 b, f32x8 c) {
  return __builtin_amdgcn_wmma_f32_16x16x32_bf16(false, a, false, b, (short)0, c, false, false);
}
// silu via hardware v_rcp_f32 (avoid IEEE divide expansion)
static __device__ __forceinline__ float silu_f(float x) {
  return x * __builtin_amdgcn_rcpf(1.0f + __expf(-x));
}
static __device__ __forceinline__ f32x8 zero8() {
  f32x8 z = {0.f,0.f,0.f,0.f,0.f,0.f,0.f,0.f}; return z;
}

// A fragment (16x32 bf16): lane m = L&15; K runs [8h..8h+7] and [16+8h..23+8h]
static __device__ __forceinline__ bf16x16 make_a_f32(const float* rowp, int half, float scale) {
  const int k0 = 8*half;
  f32x4 a0 = *(const f32x4*)(rowp + k0);
  f32x4 a1 = *(const f32x4*)(rowp + k0 + 4);
  f32x4 b0 = *(const f32x4*)(rowp + 16 + k0);
  f32x4 b1 = *(const f32x4*)(rowp + 16 + k0 + 4);
  bf16x16 r;
  #pragma unroll
  for (int i = 0; i < 4; i++) {
    r[i]    = (__bf16)(a0[i]*scale);
    r[4+i]  = (__bf16)(a1[i]*scale);
    r[8+i]  = (__bf16)(b0[i]*scale);
    r[12+i] = (__bf16)(b1[i]*scale);
  }
  return r;
}
// Same but for V tail (k_local 32..63): zero where global k>=56 (half==1 second run)
static __device__ __forceinline__ bf16x16 make_a_f32_vtail(const float* rowp, int half) {
  const int k0 = 8*half;
  f32x4 a0 = *(const f32x4*)(rowp + k0);
  f32x4 a1 = *(const f32x4*)(rowp + k0 + 4);
  f32x4 b0 = {0.f,0.f,0.f,0.f}, b1 = {0.f,0.f,0.f,0.f};
  if (half == 0) { b0 = *(const f32x4*)(rowp + 16); b1 = *(const f32x4*)(rowp + 20); }
  bf16x16 r;
  #pragma unroll
  for (int i = 0; i < 4; i++) {
    r[i]    = (__bf16)a0[i];
    r[4+i]  = (__bf16)a1[i];
    r[8+i]  = (__bf16)b0[i];
    r[12+i] = (__bf16)b1[i];
  }
  return r;
}
// A fragment from bf16 LDS hidden buffer
static __device__ __forceinline__ bf16x16 make_a_lds(const __bf16* rowp, int half) {
  bf16x8 lo = *(const bf16x8*)(rowp + 8*half);
  bf16x8 hi = *(const bf16x8*)(rowp + 16 + 8*half);
  bf16x16 r;
  #pragma unroll
  for (int i = 0; i < 8; i++) { r[i] = lo[i]; r[8+i] = hi[i]; }
  return r;
}

// ---------------- kernel 0: swizzle weights into WMMA B-fragment layout (bf16) -------------
// B frag (32x16): lane L -> col n = 16*tile + (L&15); elem e -> k = (e&7) + 8*(L>>4) + 16*(e>>3).
// Per-lane 16 contiguous bf16 (32B) => two b128 loads at use site.
__global__ __launch_bounds__(256) void prep_kernel(
    const float* __restrict__ W1r, const float* __restrict__ W2r, const float* __restrict__ W2s,
    const float* __restrict__ Wq,  const float* __restrict__ Wk,  const float* __restrict__ Wv,
    __bf16* __restrict__ w1sw, __bf16* __restrict__ w2sw,
    __bf16* __restrict__ wqsw, __bf16* __restrict__ wksw, __bf16* __restrict__ wvsw)
{
  const int tid = blockIdx.x * blockDim.x + threadIdx.x;
  const int stride = gridDim.x * blockDim.x;
  // W1r (32x112) -> 7 tiles
  for (int idx = tid; idx < 7*512; idx += stride) {
    int t = idx >> 9, rem = idx & 511, L = rem >> 4, e = rem & 15;
    int n = 16*t + (L & 15), k = (e & 7) + 8*(L >> 4) + 16*(e >> 3);
    w1sw[idx] = (__bf16)W1r[k*112 + n];
  }
  // W2cat = [W2r;W2s] (224x224) -> 7 k-steps x 14 tiles
  for (int idx = tid; idx < 7*14*512; idx += stride) {
    int s = idx / (14*512), rem = idx % (14*512);
    int t = rem >> 9, r2 = rem & 511, L = r2 >> 4, e = r2 & 15;
    int n = 16*t + (L & 15), k = 32*s + (e & 7) + 8*(L >> 4) + 16*(e >> 3);
    float v = (k < 112) ? W2r[k*224 + n] : W2s[(k - 112)*224 + n];
    w2sw[idx] = (__bf16)v;
  }
  // Wq/Wk (7,32,32): B[k][n] = W[t][n][k]
  for (int idx = tid; idx < 7*2*512; idx += stride) {
    int t = idx / 1024, rem = idx % 1024;
    int c = rem >> 9, r2 = rem & 511, L = r2 >> 4, e = r2 & 15;
    int n = 16*c + (L & 15), k = (e & 7) + 8*(L >> 4) + 16*(e >> 3);
    wqsw[idx] = (__bf16)Wq[(t*32 + n)*32 + k];
    wksw[idx] = (__bf16)Wk[(t*32 + n)*32 + k];
  }
  // Wv (4,56,56) zero-padded to 64x64: B[k][n] = Wv[h][n][k]
  for (int idx = tid; idx < 4*4*2*512; idx += stride) {
    int hh = idx / 4096, rem = idx % 4096;
    int c = rem / 1024, rem2 = rem % 1024;
    int s = rem2 >> 9, r2 = rem2 & 511, L = r2 >> 4, e = r2 & 15;
    int n = 16*c + (L & 15), k = 32*s + (e & 7) + 8*(L >> 4) + 16*(e >> 3);
    float v = (n < 56 && k < 56) ? Wv[(hh*56 + n)*56 + k] : 0.f;
    wvsw[idx] = (__bf16)v;
  }
}

// ---------------- kernel 1: node Q/K (silu, bf16) and V (f32) tables via WMMA ---------------
__global__ __launch_bounds__(256) void qkv_kernel(
    const float* __restrict__ x,
    const __bf16* __restrict__ wqsw, const __bf16* __restrict__ wksw, const __bf16* __restrict__ wvsw,
    __bf16* __restrict__ Qb, __bf16* __restrict__ Kb, float* __restrict__ Vtab, int Nn)
{
  const int lane = threadIdx.x & 31;
  const int wid  = threadIdx.x >> 5;
  const long base = (long)blockIdx.x * 128 + (long)wid * 16;
  if (base >= Nn) return;
  const int n1 = lane & 15, half = lane >> 4;
  const bool fulltile = (base + 16 <= (long)Nn);
  long arow = base + n1; if (arow > (long)Nn - 1) arow = (long)Nn - 1;
  const float* xr = x + arow * F_TOT;

  #pragma unroll 1
  for (int t = 0; t < 7; t++) {
    bf16x16 A = make_a_f32(xr + t*32, half, 1.0f);
    #pragma unroll
    for (int c = 0; c < 2; c++) {
      bf16x16 Bq = *(const bf16x16*)(wqsw + ((t*2 + c)*32 + lane)*16);
      bf16x16 Bk = *(const bf16x16*)(wksw + ((t*2 + c)*32 + lane)*16);
      f32x8 Cq = wmma_bf16(A, Bq, zero8());
      f32x8 Ck = wmma_bf16(A, Bk, zero8());
      int col = t*32 + c*16 + n1;
      if (fulltile) {
        #pragma unroll
        for (int r = 0; r < 8; r++) {
          long node = base + r + 8*half;
          Qb[node*F_TOT + col] = (__bf16)silu_f(Cq[r]);
          Kb[node*F_TOT + col] = (__bf16)silu_f(Ck[r]);
        }
      } else {
        #pragma unroll
        for (int r = 0; r < 8; r++) {
          long node = base + r + 8*half;
          if (node < Nn) {
            Qb[node*F_TOT + col] = (__bf16)silu_f(Cq[r]);
            Kb[node*F_TOT + col] = (__bf16)silu_f(Ck[r]);
          }
        }
      }
    }
  }
  #pragma unroll 1
  for (int hh = 0; hh < 4; hh++) {
    bf16x16 A0 = make_a_f32(xr + hh*56, half, 1.0f);
    bf16x16 A1 = make_a_f32_vtail(xr + hh*56 + 32, half);
    #pragma unroll 1
    for (int c = 0; c < 4; c++) {
      bf16x16 B0 = *(const bf16x16*)(wvsw + (((hh*4 + c)*2 + 0)*32 + lane)*16);
      bf16x16 B1 = *(const bf16x16*)(wvsw + (((hh*4 + c)*2 + 1)*32 + lane)*16);
      f32x8 C = wmma_bf16(A0, B0, zero8());
      C = wmma_bf16(A1, B1, C);
      int nc = c*16 + n1;
      if (nc < 56) {
        if (fulltile) {
          #pragma unroll
          for (int r = 0; r < 8; r++)
            Vtab[(base + r + 8*half)*F_TOT + hh*56 + nc] = C[r];
        } else {
          #pragma unroll
          for (int r = 0; r < 8; r++) {
            long node = base + r + 8*half;
            if (node < Nn) Vtab[node*F_TOT + hh*56 + nc] = C[r];
          }
        }
      }
    }
  }
}

// ---------------- kernel 2: edge MLP (WMMA) -> alpha -> segmented scatter ------------------
// One wave per 64-edge tile. LDS ~62.2KB.
__global__ __launch_bounds__(32) void edge_kernel(
    const float* __restrict__ rbf, const float* __restrict__ ylm, const float* __restrict__ cut,
    const int* __restrict__ idx_i, const int* __restrict__ idx_j, const float* __restrict__ ev,
    const float* __restrict__ b1r, const float* __restrict__ W1s, const float* __restrict__ b1s,
    const float* __restrict__ b2r, const float* __restrict__ b2s,
    const __bf16* __restrict__ w1sw, const __bf16* __restrict__ w2sw,
    const __bf16* __restrict__ Qb, const __bf16* __restrict__ Kb, const float* __restrict__ Vtab,
    float* __restrict__ xout, float* __restrict__ evout, int Pn)
{
  __shared__ __bf16 hbuf[64*PADW];     // hidden h = [silu(u1) | silu(u2)], bf16
  __shared__ __bf16 qkbuf[64*PADW];    // q_i * k_j product, bf16
  __shared__ float  alpha_s[64][8];
  __shared__ float  cuts[64];
  __shared__ int    iis[64], jjs[64];

  const int lane = threadIdx.x;
  const int n1 = lane & 15, half = lane >> 4;
  const long e0 = (long)blockIdx.x * 64;

  for (int i = lane; i < 64*8; i += 32) (&alpha_s[0][0])[i] = 0.0f;
  __builtin_prefetch(rbf + (e0 + 64)*32, 0, 1);   // next tile (speculative)

  // ---- per-edge setup: indices, cut, l0 contraction, qk product staging (2 edges/lane)
  float g[2][3];
  #pragma unroll 1
  for (int s = 0; s < 2; s++) {
    int el = lane + 32*s;
    long e = e0 + el;
    bool ok = (e < Pn);
    long ec = ok ? e : (long)(Pn - 1);
    int ii = idx_i[ec], jj = idx_j[ec];
    float cc = ok ? cut[ec] : 0.0f;
    iis[el] = ii; jjs[el] = jj; cuts[el] = cc;
    const float* evi = ev + (long)ii*15;
    const float* evj = ev + (long)jj*15;
    float s0 = 0.f, s1 = 0.f, s2 = 0.f;
    #pragma unroll
    for (int k2 = 0; k2 < 3;  k2++) { float d = evj[k2]-evi[k2]; s0 += d*d; }
    #pragma unroll
    for (int k2 = 3; k2 < 8;  k2++) { float d = evj[k2]-evi[k2]; s1 += d*d; }
    #pragma unroll
    for (int k2 = 8; k2 < 15; k2++) { float d = evj[k2]-evi[k2]; s2 += d*d; }
    g[s][0] = s0; g[s][1] = s1; g[s][2] = s2;
    const __bf16* qrow = Qb + (long)ii*F_TOT;
    const __bf16* krow = Kb + (long)jj*F_TOT;
    #pragma unroll 4
    for (int f = 0; f < F_TOT; f += 8) {
      bf16x8 q8 = *(const bf16x8*)(qrow + f);
      bf16x8 k8 = *(const bf16x8*)(krow + f);
      bf16x8 o;
      #pragma unroll
      for (int t = 0; t < 8; t++) o[t] = (__bf16)((float)q8[t] * (float)k8[t]);
      *(bf16x8*)(qkbuf + el*PADW + f) = o;
    }
  }

  // ---- u2 branch (K=3, VALU with uniform weight loads), hidden cols 112..223
  #pragma unroll 2
  for (int c2 = 0; c2 < 112; c2++) {
    float w0 = W1s[c2], w1 = W1s[112 + c2], w2 = W1s[224 + c2], bb = b1s[c2];
    #pragma unroll
    for (int s = 0; s < 2; s++) {
      float u = g[s][0]*w0 + g[s][1]*w1 + g[s][2]*w2 + bb;
      hbuf[(lane + 32*s)*PADW + 112 + c2] = (__bf16)silu_f(u);
    }
  }
  __syncthreads();

  // ---- layer 1 (K=32, N=112): u1 -> hidden cols 0..111
  bf16x16 A1[4];
  #pragma unroll
  for (int mt = 0; mt < 4; mt++) {
    int el = mt*16 + n1;
    long e = e0 + el; if (e > (long)Pn - 1) e = (long)Pn - 1;
    A1[mt] = make_a_f32(rbf + e*32, half, cuts[el]);   // rbf * cut
  }
  #pragma unroll 1
  for (int t7 = 0; t7 < 7; t7++) {
    bf16x16 B = *(const bf16x16*)(w1sw + (t7*32 + lane)*16);
    int col = 16*t7 + n1;
    float bb = b1r[col];
    #pragma unroll
    for (int mt = 0; mt < 4; mt++) {
      f32x8 C = wmma_bf16(A1[mt], B, zero8());
      #pragma unroll
      for (int r = 0; r < 8; r++) {
        int m = mt*16 + r + 8*half;
        hbuf[m*PADW + col] = (__bf16)silu_f(C[r] + bb);
      }
    }
  }
  __syncthreads();

  // ---- layer 2 (K=224, N=224) fused with alpha = sum_col w * (q*k)
  #pragma unroll 1
  for (int hd = 0; hd < 7; hd++) {
    float acc[4][8];
    #pragma unroll
    for (int mt = 0; mt < 4; mt++)
      #pragma unroll
      for (int r = 0; r < 8; r++) acc[mt][r] = 0.f;
    #pragma unroll 1
    for (int c = 0; c < 2; c++) {
      int nt = 2*hd + c;
      int col = 32*hd + 16*c + n1;
      float bias = b2r[col] + b2s[col];
      f32x8 C[4] = { zero8(), zero8(), zero8(), zero8() };
      #pragma unroll
      for (int ks = 0; ks < 7; ks++) {     // fully unrolled: B loads cluster, 4 indep WMMA chains
        bf16x16 B = *(const bf16x16*)(w2sw + ((ks*14 + nt)*32 + lane)*16);
        const __bf16* hb = hbuf + 32*ks;
        #pragma unroll
        for (int mt = 0; mt < 4; mt++)
          C[mt] = wmma_bf16(make_a_lds(hb + (mt*16 + n1)*PADW, half), B, C[mt]);
      }
      #pragma unroll
      for (int mt = 0; mt < 4; mt++) {
        #pragma unroll
        for (int r = 0; r < 8; r++) {
          int el = mt*16 + r + 8*half;
          float wv = C[mt][r] + bias;
          acc[mt][r] += wv * (float)qkbuf[el*PADW + col];
        }
      }
    }
    #pragma unroll
    for (int mt = 0; mt < 4; mt++)
      #pragma unroll
      for (int r = 0; r < 8; r++)
        unsafeAtomicAdd(&alpha_s[mt*16 + r + 8*half][hd], acc[mt][r]);
  }
  __syncthreads();

  // ---- scatter: segmented accumulation over sorted idx_i, one atomic per run
  #pragma unroll 1
  for (int j2 = 0; j2 < 7; j2++) {
    int f = lane + 32*j2;          // 0..223
    int hd = f / 56;               // value head 0..3
    float acc = 0.0f;
    int prev = iis[0];
    #pragma unroll 4
    for (int el = 0; el < 64; el++) {
      int node = iis[el];
      if (node != prev) {
        unsafeAtomicAdd(xout + (long)prev*F_TOT + f, acc);
        acc = 0.0f; prev = node;
      }
      float a = alpha_s[el][hd] * cuts[el];
      acc += a * Vtab[(long)jjs[el]*F_TOT + f];
    }
    unsafeAtomicAdd(xout + (long)prev*F_TOT + f, acc);
  }
  if (lane < 15) {
    int f = lane;
    int hd = 4 + ((f < 3) ? 0 : ((f < 8) ? 1 : 2));
    float acc = 0.0f;
    int prev = iis[0];
    #pragma unroll 4
    for (int el = 0; el < 64; el++) {
      long e = e0 + el; if (e > (long)Pn - 1) e = (long)Pn - 1;
      int node = iis[el];
      if (node != prev) {
        unsafeAtomicAdd(evout + (long)prev*15 + f, acc);
        acc = 0.0f; prev = node;
      }
      float a = alpha_s[el][hd] * cuts[el];
      acc += a * ylm[e*15 + f];
    }
    unsafeAtomicAdd(evout + (long)prev*15 + f, acc);
  }
}

extern "C" void kernel_launch(void* const* d_in, const int* in_sizes, int n_in,
                              void* d_out, int out_size, void* d_ws, size_t ws_size,
                              hipStream_t stream) {
  const float* x    = (const float*)d_in[0];
  const float* ev   = (const float*)d_in[1];
  const float* rbf  = (const float*)d_in[2];
  const float* ylm  = (const float*)d_in[3];
  const float* cut  = (const float*)d_in[4];
  const int*   idxi = (const int*)d_in[5];
  const int*   idxj = (const int*)d_in[6];
  const float* W1r  = (const float*)d_in[7];
  const float* b1r  = (const float*)d_in[8];
  const float* W2r  = (const float*)d_in[9];
  const float* b2r  = (const float*)d_in[10];
  const float* W1s  = (const float*)d_in[11];
  const float* b1s  = (const float*)d_in[12];
  const float* W2s  = (const float*)d_in[13];
  const float* b2s  = (const float*)d_in[14];
  const float* Wq   = (const float*)d_in[15];
  const float* Wk   = (const float*)d_in[16];
  const float* Wv   = (const float*)d_in[17];

  const int Nn = in_sizes[0] / F_TOT;
  const int Pn = in_sizes[4];

  char* ws = (char*)d_ws;
  size_t off = 0;
  auto take = [&](size_t bytes) { size_t o = off; off = (off + bytes + 255) & ~(size_t)255; return o; };
  __bf16* Qb   = (__bf16*)(ws + take((size_t)Nn * F_TOT * 2));
  __bf16* Kb   = (__bf16*)(ws + take((size_t)Nn * F_TOT * 2));
  float*  Vtab = (float*) (ws + take((size_t)Nn * F_TOT * 4));
  __bf16* w1sw = (__bf16*)(ws + take((size_t)7*512*2));
  __bf16* w2sw = (__bf16*)(ws + take((size_t)7*14*512*2));
  __bf16* wqsw = (__bf16*)(ws + take((size_t)7*2*512*2));
  __bf16* wksw = (__bf16*)(ws + take((size_t)7*2*512*2));
  __bf16* wvsw = (__bf16*)(ws + take((size_t)4*4*2*512*2));

  float* xout  = (float*)d_out;
  float* evout = (float*)d_out + (size_t)Nn * F_TOT;

  hipMemsetAsync(d_out, 0, (size_t)out_size * sizeof(float), stream);
  prep_kernel<<<64, 256, 0, stream>>>(W1r, W2r, W2s, Wq, Wk, Wv, w1sw, w2sw, wqsw, wksw, wvsw);
  qkv_kernel<<<(Nn + 127) / 128, 256, 0, stream>>>(x, wqsw, wksw, wvsw, Qb, Kb, Vtab, Nn);
  edge_kernel<<<(Pn + 63) / 64, 32, 0, stream>>>(rbf, ylm, cut, idxi, idxj, ev,
                                                 b1r, W1s, b1s, b2r, b2s,
                                                 w1sw, w2sw, Qb, Kb, Vtab,
                                                 xout, evout, Pn);
}
